// Encoder_70394513981907
// MI455X (gfx1250) — matrix operand order repaired and checked by reading, not compiled, not run on hardware
//
#include <hip/hip_runtime.h>
#include <hip/hip_bf16.h>

typedef unsigned short u16;
typedef __attribute__((ext_vector_type(8)))  __bf16 v8bf;
typedef __attribute__((ext_vector_type(16))) __bf16 v16bf;
typedef __attribute__((ext_vector_type(8)))  float  v8f;
typedef __attribute__((ext_vector_type(4)))  int    v4i;

#if __has_builtin(__builtin_amdgcn_global_load_async_to_lds_b128) && \
    __has_builtin(__builtin_amdgcn_s_wait_asynccnt)
#define USE_ASYNC_LDS 1
#endif

__device__ __forceinline__ u16 f2bf(float f) {
    return __builtin_bit_cast(u16, (__bf16)f);   // native v_cvt on gfx1250, RNE
}

__device__ __forceinline__ v8bf ldv8(const u16* p) { return *(const v8bf*)p; }

__device__ __forceinline__ v16bf cat8(v8bf lo, v8bf hi) {
    return __builtin_shufflevector(lo, hi, 0,1,2,3,4,5,6,7,8,9,10,11,12,13,14,15);
}

#ifdef USE_ASYNC_LDS
typedef __attribute__((address_space(1))) v4i* gptr_v4i;
typedef __attribute__((address_space(3))) v4i* lptr_v4i;
__device__ __forceinline__ void async_cp16(const u16* g, u16* l) {
    __builtin_amdgcn_global_load_async_to_lds_b128(
        (gptr_v4i)(u16*)g, (lptr_v4i)l, 0, 0);
}
#endif

// xor-butterfly reductions via ds_swizzle_b32 (group-of-32: imm = xor<<10 | 0x1f)
#if __has_builtin(__builtin_amdgcn_ds_swizzle)
#define SWZ_XOR(v, XM) \
    __builtin_bit_cast(float, __builtin_amdgcn_ds_swizzle(__builtin_bit_cast(int, (v)), ((XM) << 10) | 0x1f))
#else
#define SWZ_XOR(v, XM) __shfl_xor((v), (XM), 32)
#endif

__device__ __forceinline__ float red_max16(float v) {
    v = fmaxf(v, SWZ_XOR(v, 1));
    v = fmaxf(v, SWZ_XOR(v, 2));
    v = fmaxf(v, SWZ_XOR(v, 4));
    v = fmaxf(v, SWZ_XOR(v, 8));
    return v;
}
__device__ __forceinline__ float red_sum16(float v) {
    v += SWZ_XOR(v, 1);
    v += SWZ_XOR(v, 2);
    v += SWZ_XOR(v, 4);
    v += SWZ_XOR(v, 8);
    return v;
}

// ---------------------------------------------------------------------------
// Weight transpose + fp32 -> bf16 convert:  W[K][N] -> Wt[N][K] (bf16)
// ---------------------------------------------------------------------------
__global__ __launch_bounds__(256)
void tconv(const float* __restrict__ W, u16* __restrict__ Wt, int K, int N) {
    __shared__ float t[32][33];
    const int k0 = blockIdx.y * 32, n0 = blockIdx.x * 32;
    const int tx = threadIdx.x, ty = threadIdx.y;       // 32 x 8
#pragma unroll
    for (int i = 0; i < 4; ++i)
        t[ty + i * 8][tx] = W[(size_t)(k0 + ty + i * 8) * N + n0 + tx];
    __syncthreads();
#pragma unroll
    for (int i = 0; i < 4; ++i)
        Wt[(size_t)(n0 + ty + i * 8) * K + k0 + tx] = f2bf(t[tx][ty + i * 8]);
}

// ---------------------------------------------------------------------------
// LayerNorm (D=768) -> bf16 row
// ---------------------------------------------------------------------------
__global__ __launch_bounds__(256)
void ln_bf16(const float* __restrict__ x, const float* __restrict__ g,
             const float* __restrict__ b, u16* __restrict__ out) {
    __shared__ float red[256];
    const int row = blockIdx.x;
    const float* xr = x + (size_t)row * 768;
    float s = 0.f;
    for (int i = threadIdx.x; i < 768; i += 256) s += xr[i];
    red[threadIdx.x] = s; __syncthreads();
    for (int o = 128; o > 0; o >>= 1) {
        if (threadIdx.x < o) red[threadIdx.x] += red[threadIdx.x + o];
        __syncthreads();
    }
    const float mu = red[0] * (1.f / 768.f);
    __syncthreads();
    float vs = 0.f;
    for (int i = threadIdx.x; i < 768; i += 256) { float d = xr[i] - mu; vs += d * d; }
    red[threadIdx.x] = vs; __syncthreads();
    for (int o = 128; o > 0; o >>= 1) {
        if (threadIdx.x < o) red[threadIdx.x] += red[threadIdx.x + o];
        __syncthreads();
    }
    const float rs = rsqrtf(red[0] * (1.f / 768.f) + 1e-6f);
    for (int i = threadIdx.x; i < 768; i += 256)
        out[(size_t)row * 768 + i] = f2bf((xr[i] - mu) * rs * g[i] + b[i]);
}

// ---------------------------------------------------------------------------
// bf16 GEMM, C = A[M][K] * Bt[N][K]^T, fp32 accum via v_wmma_f32_16x16x32_bf16
// Block tile 128x128x32, 8 waves (4x2) of 32x64, double-buffered LDS with
// async global->LDS DMA staging (ASYNCcnt) when available.
// MODE 0: +bias, scatter bf16 to qkv [3][B][H][S][64]
// MODE 1: +bias +resid -> fp32
// MODE 2: +bias, exact GELU -> bf16
// MODE 3: +bias +resid -> fp32 (final output)
// ---------------------------------------------------------------------------
template <int MODE>
__global__ __launch_bounds__(256)
void gemm_bf16(const u16* __restrict__ A, const u16* __restrict__ Bt,
               const float* __restrict__ bias, const float* __restrict__ resid,
               float* __restrict__ outF, u16* __restrict__ outB,
               int M, int N, int K) {
    __shared__ u16 sA[2][128][40];   // stride 40 ushorts = 80B (16B aligned, bank-spread)
    __shared__ u16 sB[2][128][40];
    const int tid = threadIdx.x;
    const int lane = tid & 31, w = tid >> 5;
    const int wm = w & 3, wn = w >> 2;
    const int hi = lane >> 4, ln = lane & 15;
    const int m0 = blockIdx.y * 128, n0 = blockIdx.x * 128;
    const int KT = K >> 5;

    v8f acc[2][4];
#pragma unroll
    for (int mi = 0; mi < 2; ++mi)
#pragma unroll
        for (int ni = 0; ni < 4; ++ni)
#pragma unroll
            for (int r = 0; r < 8; ++r) acc[mi][ni][r] = 0.f;

    auto stage = [&](int kt, int buf) {
        const int kb = kt << 5;
#pragma unroll
        for (int i = 0; i < 2; ++i) {
            const int c = tid + i * 256;          // 512 16B-chunks total
            const int r = c >> 2, c8 = (c & 3) << 3;
#ifdef USE_ASYNC_LDS
            async_cp16(A + (size_t)(m0 + r) * K + kb + c8, &sA[buf][r][c8]);
            async_cp16(Bt + (size_t)(n0 + r) * K + kb + c8, &sB[buf][r][c8]);
#else
            uint4 a = *(const uint4*)(A + (size_t)(m0 + r) * K + kb + c8);
            *(uint4*)(&sA[buf][r][c8]) = a;
            uint4 bb = *(const uint4*)(Bt + (size_t)(n0 + r) * K + kb + c8);
            *(uint4*)(&sB[buf][r][c8]) = bb;
#endif
        }
    };

    stage(0, 0);
#ifdef USE_ASYNC_LDS
    __builtin_amdgcn_s_wait_asynccnt(0);
#endif
    __syncthreads();
    for (int kt = 0; kt < KT; ++kt) {
        const int cur = kt & 1;
        if (kt + 1 < KT) stage(kt + 1, cur ^ 1);
        if (kt + 2 < KT) {
            __builtin_prefetch(A + (size_t)(m0 + (tid >> 2)) * K + ((kt + 2) << 5), 0, 0);
            __builtin_prefetch(Bt + (size_t)(n0 + (tid >> 2)) * K + ((kt + 2) << 5), 0, 0);
        }
        // A frag: lane = M row, elems 0..7 -> K = hi*8.., elems 8..15 -> +16
        v16bf af[2];
#pragma unroll
        for (int mi = 0; mi < 2; ++mi) {
            const u16* p = &sA[cur][wm * 32 + mi * 16 + ln][hi * 8];
            af[mi] = cat8(ldv8(p), ldv8(p + 16));
        }
        // B frag: lane = N col, elems 0..7 -> K = hi*8.., elems 8..15 -> +16
        v16bf bw[4];
#pragma unroll
        for (int ni = 0; ni < 4; ++ni) {
            const u16* p = &sB[cur][wn * 64 + ni * 16 + ln][hi * 8];
            bw[ni] = cat8(ldv8(p), ldv8(p + 16));
        }
#pragma unroll
        for (int mi = 0; mi < 2; ++mi)
#pragma unroll
            for (int ni = 0; ni < 4; ++ni)
                acc[mi][ni] = __builtin_amdgcn_wmma_f32_16x16x32_bf16(
                    false, af[mi], false, bw[ni], (short)0, acc[mi][ni], false, false);
#ifdef USE_ASYNC_LDS
        if (kt + 1 < KT) __builtin_amdgcn_s_wait_asynccnt(0);
#endif
        __syncthreads();
    }

    // Epilogue. C layout: VGPR r -> row r (+8 for upper lane half), lanes = cols.
#pragma unroll
    for (int mi = 0; mi < 2; ++mi)
#pragma unroll
        for (int ni = 0; ni < 4; ++ni)
#pragma unroll
            for (int r = 0; r < 8; ++r) {
                const int row = m0 + wm * 32 + mi * 16 + r + hi * 8;
                const int col = n0 + wn * 64 + ni * 16 + ln;
                float v = acc[mi][ni][r] + bias[col];
                if (MODE == 0) {
                    const int which = col / 768, rem = col - which * 768;
                    const int hh = rem >> 6, d = rem & 63;
                    const int bb = row >> 11, s = row & 2047;
                    const size_t di = ((((size_t)which * 2 + bb) * 12 + hh) * 2048 + s) * 64 + d;
                    outB[di] = f2bf(v);
                } else if (MODE == 1 || MODE == 3) {
                    const size_t di = (size_t)row * N + col;
                    outF[di] = v + resid[di];
                } else {
                    const float ge = 0.5f * v * (1.0f + erff(v * 0.7071067811865476f));
                    outB[(size_t)row * N + col] = f2bf(ge);
                }
            }
}

// ---------------------------------------------------------------------------
// Flash attention: one block = one (b,h) x 128 Q rows; 8 waves x 16 rows.
// Q/K frags straight from global (layouts match ISA A/B striping),
// V transposed through LDS, P through per-wave LDS, online softmax with
// ds_swizzle xor-butterfly row reductions.
// ---------------------------------------------------------------------------
__global__ __launch_bounds__(256)
void flash_attn(const u16* __restrict__ Q, const u16* __restrict__ Kb,
                const u16* __restrict__ Vb, u16* __restrict__ out) {
    __shared__ u16 sV[64][72];        // V^T tile: [hd][t], stride 72 (144B, 16B aligned)
    __shared__ u16 sP[8][16][72];     // per-wave P staging
    const int tid = threadIdx.x, lane = tid & 31, w = tid >> 5;
    const int hi = lane >> 4, ln = lane & 15;
    const int h = blockIdx.y, b = blockIdx.z;
    const size_t bh = ((size_t)b * 12 + h) * 2048;
    const u16* qp = Q + bh * 64;
    const u16* kp = Kb + bh * 64;
    const u16* vp = Vb + bh * 64;
    const int q0 = blockIdx.x * 128 + w * 16;

    v16bf aq[2];
#pragma unroll
    for (int kc = 0; kc < 2; ++kc) {
        const u16* p = qp + (size_t)(q0 + ln) * 64 + kc * 32 + hi * 8;
        aq[kc] = cat8(ldv8(p), ldv8(p + 16));
    }

    float run_m[8], run_s[8];
    v8f o[4];
#pragma unroll
    for (int r = 0; r < 8; ++r) { run_m[r] = -1e30f; run_s[r] = 0.f; }
#pragma unroll
    for (int ni = 0; ni < 4; ++ni)
#pragma unroll
        for (int r = 0; r < 8; ++r) o[ni][r] = 0.f;

    for (int t = 0; t < 32; ++t) {
        __syncthreads();                              // protect sV from prev readers
#pragma unroll
        for (int i = 0; i < 2; ++i) {                 // stage V^T tile
            const int c = tid + i * 256;
            const int tr = c >> 3, d0 = (c & 7) << 3;
            v8bf vv = ldv8(vp + (size_t)(t * 64 + tr) * 64 + d0);
#pragma unroll
            for (int j = 0; j < 8; ++j) sV[d0 + j][tr] = ((const u16*)&vv)[j];
        }
        __syncthreads();
        if (t + 1 < 32) {
            __builtin_prefetch(vp + (size_t)((t + 1) * 64 + (tid >> 2)) * 64, 0, 0);
            __builtin_prefetch(kp + (size_t)((t + 1) * 64 + (tid >> 2)) * 64, 0, 0);
        }

        // scores S = Q K^T : B[k][n] = K[row n][k] -> contiguous per lane in global
        v8f sc[4];
#pragma unroll
        for (int nb = 0; nb < 4; ++nb) {
#pragma unroll
            for (int r = 0; r < 8; ++r) sc[nb][r] = 0.f;
#pragma unroll
            for (int kc = 0; kc < 2; ++kc) {
                const u16* p = kp + (size_t)(t * 64 + nb * 16 + ln) * 64 + kc * 32 + hi * 8;
                v16bf bk = cat8(ldv8(p), ldv8(p + 16));
                sc[nb] = __builtin_amdgcn_wmma_f32_16x16x32_bf16(
                    false, aq[kc], false, bk, (short)0, sc[nb], false, false);
            }
        }
#pragma unroll
        for (int nb = 0; nb < 4; ++nb)
#pragma unroll
            for (int r = 0; r < 8; ++r) sc[nb][r] *= 0.125f;   // 1/sqrt(64)

        // online softmax: VGPR r = row, 16-lane halves hold its 16 cols
#pragma unroll
        for (int r = 0; r < 8; ++r) {
            float m = fmaxf(fmaxf(sc[0][r], sc[1][r]), fmaxf(sc[2][r], sc[3][r]));
            m = red_max16(m);
            const float nm = fmaxf(run_m[r], m);
            const float corr = __expf(run_m[r] - nm);
            float rs = 0.f;
#pragma unroll
            for (int nb = 0; nb < 4; ++nb) {
                const float pe = __expf(sc[nb][r] - nm);
                sc[nb][r] = pe; rs += pe;
            }
            rs = red_sum16(rs);
            run_s[r] = run_s[r] * corr + rs;
            run_m[r] = nm;
#pragma unroll
            for (int ni = 0; ni < 4; ++ni) o[ni][r] *= corr;
        }

        // C-layout -> A-layout through per-wave LDS (same-wave, DScnt-ordered)
#pragma unroll
        for (int nb = 0; nb < 4; ++nb)
#pragma unroll
            for (int r = 0; r < 8; ++r)
                sP[w][r + hi * 8][nb * 16 + ln] = f2bf(sc[nb][r]);

        v16bf ap[2];
#pragma unroll
        for (int kc = 0; kc < 2; ++kc) {
            const u16* p = &sP[w][ln][kc * 32 + hi * 8];
            ap[kc] = cat8(ldv8(p), ldv8(p + 16));
        }
        // O += P V : B[k][n] = V[k][n] = sV[n][k], k = hi*8.. and +16 per lane
#pragma unroll
        for (int ni = 0; ni < 4; ++ni)
#pragma unroll
            for (int kc = 0; kc < 2; ++kc) {
                const u16* p = &sV[ni * 16 + ln][kc * 32 + hi * 8];
                v16bf bv = cat8(ldv8(p), ldv8(p + 16));
                o[ni] = __builtin_amdgcn_wmma_f32_16x16x32_bf16(
                    false, ap[kc], false, bv, (short)0, o[ni], false, false);
            }
    }

#pragma unroll
    for (int r = 0; r < 8; ++r) {
        const float inv = 1.0f / run_s[r];
        const int s = q0 + r + hi * 8;
#pragma unroll
        for (int ni = 0; ni < 4; ++ni)
            out[((size_t)b * 2048 + s) * 768 + h * 64 + ni * 16 + ln] =
                f2bf(o[ni][r] * inv);
    }
}

// ---------------------------------------------------------------------------
extern "C" void kernel_launch(void* const* d_in, const int* in_sizes, int n_in,
                              void* d_out, int out_size, void* d_ws, size_t ws_size,
                              hipStream_t stream) {
    const float* src  = (const float*)d_in[0];
    const float* ln1g = (const float*)d_in[1];
    const float* ln1b = (const float*)d_in[2];
    const float* Wqkv = (const float*)d_in[3];
    const float* bqkv = (const float*)d_in[4];
    const float* Wo   = (const float*)d_in[5];
    const float* bo   = (const float*)d_in[6];
    const float* ln2g = (const float*)d_in[7];
    const float* ln2b = (const float*)d_in[8];
    const float* W1   = (const float*)d_in[9];
    const float* b1   = (const float*)d_in[10];
    const float* W2   = (const float*)d_in[11];
    const float* b2   = (const float*)d_in[12];
    float* out = (float*)d_out;

    char* ws = (char*)d_ws;
    size_t off = 0;
    auto alloc = [&](size_t bytes) {
        void* p = ws + off;
        off = (off + bytes + 255) & ~(size_t)255;
        return p;
    };
    u16*   Xa   = (u16*)  alloc(4096ull * 768 * 2);
    u16*   Wqt  = (u16*)  alloc(2304ull * 768 * 2);
    u16*   Wot  = (u16*)  alloc(768ull * 768 * 2);
    u16*   W1t  = (u16*)  alloc(3072ull * 768 * 2);
    u16*   W2t  = (u16*)  alloc(768ull * 3072 * 2);
    u16*   qkv  = (u16*)  alloc(3ull * 4096 * 768 * 2);
    u16*   attn = (u16*)  alloc(4096ull * 768 * 2);
    float* src2 = (float*)alloc(4096ull * 768 * 4);
    u16*   Xb   = (u16*)  alloc(4096ull * 768 * 2);
    u16*   Hb   = (u16*)  alloc(4096ull * 3072 * 2);

    const dim3 tb(32, 8);
    tconv<<<dim3(2304 / 32, 768 / 32),  tb, 0, stream>>>(Wqkv, Wqt, 768, 2304);
    tconv<<<dim3(768 / 32,  768 / 32),  tb, 0, stream>>>(Wo,   Wot, 768, 768);
    tconv<<<dim3(3072 / 32, 768 / 32),  tb, 0, stream>>>(W1,   W1t, 768, 3072);
    tconv<<<dim3(768 / 32,  3072 / 32), tb, 0, stream>>>(W2,   W2t, 3072, 768);

    ln_bf16<<<4096, 256, 0, stream>>>(src, ln1g, ln1b, Xa);

    gemm_bf16<0><<<dim3(2304 / 128, 4096 / 128), 256, 0, stream>>>(
        Xa, Wqt, bqkv, nullptr, nullptr, qkv, 4096, 2304, 768);

    const u16* Qp = qkv;
    const u16* Kp = qkv + 4096ull * 768;
    const u16* Vp = qkv + 2ull * 4096 * 768;
    flash_attn<<<dim3(16, 12, 2), 256, 0, stream>>>(Qp, Kp, Vp, attn);

    gemm_bf16<1><<<dim3(768 / 128, 4096 / 128), 256, 0, stream>>>(
        attn, Wot, bo, src, src2, nullptr, 4096, 768, 768);

    ln_bf16<<<4096, 256, 0, stream>>>(src2, ln2g, ln2b, Xb);

    gemm_bf16<2><<<dim3(3072 / 128, 4096 / 128), 256, 0, stream>>>(
        Xb, W1t, b1, nullptr, nullptr, Hb, 4096, 3072, 768);

    gemm_bf16<3><<<dim3(768 / 128, 4096 / 128), 256, 0, stream>>>(
        Hb, W2t, b2, src2, out, nullptr, 4096, 768, 3072);
}
